// FusedFP8Linear_34333968564605
// MI455X (gfx1250) — 8-bit matrix operand order repaired and checked by reading, compiled, not run on hardware
//
#include <hip/hip_runtime.h>
#include <hip/hip_bf16.h>

typedef __attribute__((ext_vector_type(16))) int   v16i;
typedef __attribute__((ext_vector_type(8)))  float v8f;
typedef __attribute__((ext_vector_type(4)))  int   v4i;

#define AS_GLOBAL __attribute__((address_space(1)))
#define AS_LDS    __attribute__((address_space(3)))

#if defined(__has_builtin)
#  if __has_builtin(__builtin_amdgcn_global_load_async_to_lds_b128)
#    define HAVE_ASYNC_LDS 1
#  endif
#  if __has_builtin(__builtin_amdgcn_s_wait_asynccnt)
#    define HAVE_WAIT_ASYNC 1
#  endif
#  if __has_builtin(__builtin_amdgcn_cvt_pk_fp8_f32)
#    define HAVE_CVT_FP8 1
#  endif
#endif

__device__ __forceinline__ void copy16_g2l(const unsigned char* g, unsigned char* l) {
#if defined(HAVE_ASYNC_LDS)
  __builtin_amdgcn_global_load_async_to_lds_b128((AS_GLOBAL v4i*)g, (AS_LDS v4i*)l,
                                                  0,  0);
#else
  *(uint4*)l = *(const uint4*)g;
#endif
}

__device__ __forceinline__ void wait_async_all() {
#if defined(HAVE_ASYNC_LDS)
#  if defined(HAVE_WAIT_ASYNC)
  __builtin_amdgcn_s_wait_asynccnt(0);
#  else
  asm volatile("s_wait_asynccnt 0x0" ::: "memory");
#  endif
#endif
}

__device__ __forceinline__ unsigned int f32_to_e4m3_sw(float x) {
  const unsigned int s = (__float_as_uint(x) >> 31) << 7;
  float ax = fabsf(x);
  ax = fminf(ax, 448.0f);
  if (!(ax >= 0.0009765625f * 0.125f))
    return s;
  unsigned int bits = __float_as_uint(ax);
  int e = (int)(bits >> 23) - 127;
  if (e < -6) {
    float r = rintf(ax * 512.0f);
    return s | (unsigned int)r;
  }
  const float sc = __uint_as_float((unsigned int)(127 + 3 - e) << 23);
  int q = (int)rintf(ax * sc);
  if (q == 16) { q = 8; e += 1; }
  if (e > 8)   { e = 8; q = 15; }
  return s | ((unsigned int)(e + 7) << 3) | (unsigned int)(q - 8);
}

__global__ void __launch_bounds__(256)
quant_bf16_e4m3(const __hip_bfloat16* __restrict__ src,
                unsigned char* __restrict__ dst,
                const float* __restrict__ scale_p,
                long long n8) {
  long long i = (long long)blockIdx.x * blockDim.x + threadIdx.x;
  if (i >= n8) return;
  const float inv = 1.0f / scale_p[0];
  const uint4 raw = ((const uint4* __restrict__)src)[i];
  const unsigned int w[4] = {raw.x, raw.y, raw.z, raw.w};
  float f[8];
#pragma unroll
  for (int j = 0; j < 4; ++j) {
    f[2 * j]     = __uint_as_float((w[j] & 0x0000ffffu) << 16) * inv;
    f[2 * j + 1] = __uint_as_float(w[j] & 0xffff0000u) * inv;
  }
#pragma unroll
  for (int j = 0; j < 8; ++j) f[j] = fminf(fmaxf(f[j], -448.0f), 448.0f);
#if defined(HAVE_CVT_FP8)
  int lo = __builtin_amdgcn_cvt_pk_fp8_f32(f[0], f[1], 0, false);
  lo     = __builtin_amdgcn_cvt_pk_fp8_f32(f[2], f[3], lo, true);
  int hi = __builtin_amdgcn_cvt_pk_fp8_f32(f[4], f[5], 0, false);
  hi     = __builtin_amdgcn_cvt_pk_fp8_f32(f[6], f[7], hi, true);
  ((uint2*)dst)[i] = make_uint2((unsigned int)lo, (unsigned int)hi);
#else
  unsigned int lo = 0u, hi = 0u;
#pragma unroll
  for (int j = 0; j < 4; ++j) lo |= f32_to_e4m3_sw(f[j]) << (8 * j);
#pragma unroll
  for (int j = 0; j < 4; ++j) hi |= f32_to_e4m3_sw(f[4 + j]) << (8 * j);
  ((uint2*)dst)[i] = make_uint2(lo, hi);
#endif
}

#define BM 128
#define BN 128
#define BK 128
#define LDS_ROW 144

__global__ void __launch_bounds__(256)
gemm_fp8_wmma(const unsigned char* __restrict__ xq,
              const unsigned char* __restrict__ wq,
              const __hip_bfloat16* __restrict__ bias,
              const float* __restrict__ xs_p,
              const float* __restrict__ ws_p,
              __hip_bfloat16* __restrict__ out,
              int M, int N, int K) {
  __shared__ unsigned char As[2][BM * LDS_ROW];
  __shared__ unsigned char Bs[2][BN * LDS_ROW];

  const int t     = threadIdx.x;
  const int lane  = t & 31;
  const int wave  = t >> 5;
  const int waveM = wave >> 2;
  const int waveN = wave & 3;
  const int lhalf = lane >> 4;
  const int l16   = lane & 15;

  const int rowBlock = blockIdx.y * BM;
  const int colBlock = blockIdx.x * BN;

  int cOff[4];
  const unsigned char* aG[4];
  const unsigned char* bG[4];
#pragma unroll
  for (int j = 0; j < 4; ++j) {
    const int c   = t + 256 * j;
    const int row = c >> 3;
    const int col = (c & 7) * 16;
    cOff[j] = row * LDS_ROW + col;
    aG[j] = xq + (size_t)(rowBlock + row) * K + col;
    bG[j] = wq + (size_t)(colBlock + row) * K + col;
  }

  v8f acc[4][2];
#pragma unroll
  for (int mt = 0; mt < 4; ++mt)
#pragma unroll
    for (int nt = 0; nt < 2; ++nt)
#pragma unroll
      for (int i = 0; i < 8; ++i) acc[mt][nt][i] = 0.0f;

#pragma unroll
  for (int j = 0; j < 4; ++j) {
    copy16_g2l(aG[j], &As[0][cOff[j]]);
    copy16_g2l(bG[j], &Bs[0][cOff[j]]);
  }

  int p = 0;
  for (int kk = 0; kk < K; kk += BK) {
    wait_async_all();
    __syncthreads();

    if (kk + BK < K) {
      const int kn = kk + BK;
#pragma unroll
      for (int j = 0; j < 4; ++j) {
        copy16_g2l(aG[j] + kn, &As[p ^ 1][cOff[j]]);
        copy16_g2l(bG[j] + kn, &Bs[p ^ 1][cOff[j]]);
      }
    }

    union { v16i v; unsigned long long d[8]; } a[4];
    union { v16i v; uint4 q[4]; } b[2];
    const unsigned char* aT = &As[p][0];
    const unsigned char* bT = &Bs[p][0];
#pragma unroll
    for (int mt = 0; mt < 4; ++mt) {
      const unsigned char* r = aT + (waveM * 64 + mt * 16 + l16) * LDS_ROW + (lhalf << 4);
#pragma unroll
      for (int j = 0; j < 8; ++j)
        a[mt].d[j] = *(const unsigned long long*)(r + (j >> 1) * 32 + (j & 1) * 8);
    }
#pragma unroll
    for (int nt = 0; nt < 2; ++nt) {
      const unsigned char* r = bT + (waveN * 32 + nt * 16 + l16) * LDS_ROW + (lhalf << 4);
#pragma unroll
      for (int j = 0; j < 4; ++j)
        b[nt].q[j] = *(const uint4*)(r + j * 32);
    }

#pragma unroll
    for (int mt = 0; mt < 4; ++mt)
#pragma unroll
      for (int nt = 0; nt < 2; ++nt)
        acc[mt][nt] = __builtin_amdgcn_wmma_f32_16x16x128_fp8_fp8(
            a[mt].v, b[nt].v, (short)0, acc[mt][nt], false, false);

    p ^= 1;
  }

  const float sc = xs_p[0] * ws_p[0];
#pragma unroll
  for (int nt = 0; nt < 2; ++nt) {
    const int n = colBlock + waveN * 32 + nt * 16 + l16;
    const float bn = __bfloat162float(bias[n]);
#pragma unroll
    for (int mt = 0; mt < 4; ++mt) {
      const int mtop = rowBlock + waveM * 64 + mt * 16 + (lhalf ? 8 : 0);
#pragma unroll
      for (int i = 0; i < 8; ++i) {
        const int m = mtop + i;
        out[(size_t)m * N + n] = __float2bfloat16(acc[mt][nt][i] * sc + bn);
      }
    }
  }
}

extern "C" void kernel_launch(void* const* d_in, const int* in_sizes, int n_in,
                              void* d_out, int out_size, void* d_ws, size_t ws_size,
                              hipStream_t stream) {
  const __hip_bfloat16* x    = (const __hip_bfloat16*)d_in[0];
  const __hip_bfloat16* wgt  = (const __hip_bfloat16*)d_in[1];
  const __hip_bfloat16* bias = (const __hip_bfloat16*)d_in[2];
  const float* xs = (const float*)d_in[3];
  const float* ws = (const float*)d_in[4];

  const int       N  = in_sizes[2];
  const long long NK = (long long)in_sizes[1];
  const int       K  = (int)(NK / N);
  const long long MK = (long long)in_sizes[0];
  const int       M  = (int)(MK / K);

  unsigned char* xq = (unsigned char*)d_ws;
  unsigned char* wq = xq + (size_t)M * (size_t)K;

  {
    const long long n8x = MK / 8;
    quant_bf16_e4m3<<<(unsigned)((n8x + 255) / 256), 256, 0, stream>>>(x, xq, xs, n8x);
    const long long n8w = NK / 8;
    quant_bf16_e4m3<<<(unsigned)((n8w + 255) / 256), 256, 0, stream>>>(wgt, wq, ws, n8w);
  }

  dim3 grid((unsigned)(N / BN), (unsigned)(M / BM));
  gemm_fp8_wmma<<<grid, 256, 0, stream>>>(xq, wq, bias, xs, ws,
                                          (__hip_bfloat16*)d_out, M, N, K);
}
